// FAVORPlusAttention_88416196755756
// MI455X (gfx1250) — matrix operand order repaired and checked by reading, not compiled, not run on hardware
//
#include <hip/hip_runtime.h>
#include <hip/hip_bf16.h>
#include <math.h>
#include <stdint.h>

typedef __bf16 bf16;
typedef __attribute__((ext_vector_type(16))) __bf16 v16bf;
typedef __attribute__((ext_vector_type(8)))  float  v8f;
typedef unsigned int v4u __attribute__((ext_vector_type(4)));
typedef unsigned int v8u __attribute__((ext_vector_type(8)));

#define D_MODEL 1024
#define N_HEADS 16
#define HEAD_DIM 64
#define N_FEAT 256
#define SEQ 4096
#define BATCH 4
#define ROWS (BATCH*SEQ)   // 16384
#define EPSF 1e-6f

// ---------------------------------------------------------------------------
// CDNA5 async copy: global -> LDS, tracked by ASYNCcnt (no VGPR round-trip)
// ---------------------------------------------------------------------------
static __device__ __forceinline__ void async_ld_b128(const void* g, void* l) {
  asm volatile("global_load_async_to_lds_b128 %0, %1, off"
               :: "v"((uint32_t)(uintptr_t)l), "v"(g)
               : "memory");
}
static __device__ __forceinline__ void wait_async0() {
  asm volatile("s_wait_asynccnt 0" ::: "memory");
}

// ---------------------------------------------------------------------------
// WMMA fragment helpers (wave32, 16x16x32 bf16, fp32 accum)
// A 16x32: lane m=lane&15, half=lane>>4; vec elem j: k = half*16 + j
// B 32x16 from Bt (=B^T, row-major [N][K]): lane n=lane&15; k = half*16 + i
// C/D 16x16: row = half*8 + r, col = n
// ---------------------------------------------------------------------------
static __device__ __forceinline__ v16bf frag_a_ld(const bf16* p, int ld) {
  int lane = threadIdx.x & 31;
  int m = lane & 15, half = lane >> 4;
  v16bf r;
#pragma unroll
  for (int g = 0; g < 2; ++g) {
    const bf16* q = p + (size_t)m * ld + half * 16;
#pragma unroll
    for (int i = 0; i < 8; ++i) r[g * 8 + i] = q[g * 8 + i];
  }
  return r;
}

static __device__ __forceinline__ v16bf frag_b_ld(const bf16* pt, int ld) {
  int lane = threadIdx.x & 31;
  int n = lane & 15, half = lane >> 4;
  const bf16* q = pt + (size_t)n * ld + half * 16;
  v16bf r;
#pragma unroll
  for (int i = 0; i < 16; ++i) r[i] = q[i];
  return r;
}

static __device__ __forceinline__ v16bf frag_b_ld_f32(const float* pt, int ld) {
  int lane = threadIdx.x & 31;
  int n = lane & 15, half = lane >> 4;
  const float* q = pt + (size_t)n * ld + half * 16;
  v16bf r;
#pragma unroll
  for (int i = 0; i < 16; ++i) r[i] = (bf16)q[i];
  return r;
}

static __device__ __forceinline__ v8f wmma_bf16(v16bf a, v16bf b, v8f c) {
  return __builtin_amdgcn_wmma_f32_16x16x32_bf16(false, a, false, b, (short)0, c,
                                                 false, false);
}

// ---------------------------------------------------------------------------
// fp32 -> bf16 conversion / transpose-conversion
// ---------------------------------------------------------------------------
__global__ void k_cvt(const float* __restrict__ s, bf16* __restrict__ d, size_t n) {
  size_t i = (size_t)blockIdx.x * blockDim.x + threadIdx.x;
  if (i < n) d[i] = (bf16)s[i];
}

// Wt[n*K + k] = W[k*N + n]
__global__ void k_transpose_cvt(const float* __restrict__ W, bf16* __restrict__ Wt,
                                int K, int N) {
  size_t i = (size_t)blockIdx.x * blockDim.x + threadIdx.x;
  if (i < (size_t)K * N) {
    int n = (int)(i % N);
    int k = (int)(i / N);
    Wt[(size_t)n * K + k] = (bf16)W[i];
  }
}

// ---------------------------------------------------------------------------
// Generic bf16 GEMM: C[M,N] = (A[M,K] * Bt[N,K]^T + bias) * scale
// Block = 64x64 output tile, 256 threads (8 waves), 2 WMMA tiles per wave.
// Staging via GLOBAL_LOAD_ASYNC_TO_LDS_B128.
// ---------------------------------------------------------------------------
__global__ void __launch_bounds__(256)
k_gemm(const bf16* __restrict__ A, const bf16* __restrict__ Bt,
       const float* __restrict__ bias, float* __restrict__ outF,
       bf16* __restrict__ outB, int M, int N, int K, float scale) {
  __shared__ bf16 As[64][32];
  __shared__ bf16 Bs[64][32];
  int tid = threadIdx.x, w = tid >> 5, lane = tid & 31;
  int n0 = blockIdx.x * 64, m0 = blockIdx.y * 64;
  int tt0 = w, tt1 = w + 8;
  int ti0 = tt0 >> 2, tj0 = tt0 & 3, ti1 = tt1 >> 2, tj1 = tt1 & 3;
  int lrow = tid >> 2;            // 0..63
  int lcol = (tid & 3) * 8;       // 0,8,16,24 (bf16 elems -> 16B each)
  v8f acc0 = {}, acc1 = {};

  for (int k0 = 0; k0 < K; k0 += 32) {
    __syncthreads();
    async_ld_b128(A + (size_t)(m0 + lrow) * K + k0 + lcol, &As[lrow][lcol]);
    async_ld_b128(Bt + (size_t)(n0 + lrow) * K + k0 + lcol, &Bs[lrow][lcol]);
    if (k0 + 32 < K) {
      __builtin_prefetch(A + (size_t)(m0 + lrow) * K + k0 + 32 + lcol, 0, 3);
      __builtin_prefetch(Bt + (size_t)(n0 + lrow) * K + k0 + 32 + lcol, 0, 3);
    }
    wait_async0();
    __syncthreads();
    v16bf a0 = frag_a_ld(&As[ti0 * 16][0], 32);
    v16bf b0 = frag_b_ld(&Bs[tj0 * 16][0], 32);
    acc0 = wmma_bf16(a0, b0, acc0);
    v16bf a1 = frag_a_ld(&As[ti1 * 16][0], 32);
    v16bf b1 = frag_b_ld(&Bs[tj1 * 16][0], 32);
    acc1 = wmma_bf16(a1, b1, acc1);
  }

  int n = lane & 15, half = lane >> 4;
#pragma unroll
  for (int t = 0; t < 2; ++t) {
    int ti = t ? ti1 : ti0, tj = t ? tj1 : tj0;
    v8f acc = t ? acc1 : acc0;
    int col = n0 + tj * 16 + n;
    float bv = bias ? bias[col] : 0.f;
#pragma unroll
    for (int r = 0; r < 8; ++r) {
      int row = m0 + ti * 16 + half * 8 + r;
      float v = (acc[r] + bv) * scale;
      if (outF) outF[(size_t)row * N + col] = v;
      if (outB) outB[(size_t)row * N + col] = (bf16)v;
    }
  }
}

// ---------------------------------------------------------------------------
// phi kernel: P[bh, l, m] = exp( xs @ omega^T - 0.5*||xs||^2 ) / sqrt(M)
// Xs already scaled by HEAD_DIM^-0.25 and stored bf16 [ROWS, D_MODEL].
// One block = 64 rows of one (b,h); output 64x256 = 64 WMMA tiles (8/wave).
// ---------------------------------------------------------------------------
__global__ void __launch_bounds__(256)
k_phi(const bf16* __restrict__ Xs, const bf16* __restrict__ Om,
      bf16* __restrict__ P) {
  __shared__ bf16 As[64][64];      // 8 KB
  __shared__ bf16 Os[256][64];     // 32 KB  (omega, row-major = Bt)
  __shared__ float norms[64];
  int tid = threadIdx.x, w = tid >> 5, lane = tid & 31;
  int bh = blockIdx.y;
  int b = bh >> 4, h = bh & 15;
  int row0 = blockIdx.x * 64;
  const bf16* Abase = Xs + ((size_t)(b * SEQ + row0)) * D_MODEL + h * HEAD_DIM;

  for (int q = tid; q < 512; q += 256) {            // 64*64 bf16 / 8
    int row = q >> 3, col = (q & 7) * 8;
    async_ld_b128(Abase + (size_t)row * D_MODEL + col, &As[row][col]);
  }
  for (int q = tid; q < 2048; q += 256) {           // 256*64 bf16 / 8
    int row = q >> 3, col = (q & 7) * 8;
    async_ld_b128(Om + (size_t)row * 64 + col, &Os[row][col]);
  }
  wait_async0();
  __syncthreads();
  if (tid < 64) {
    float s = 0.f;
#pragma unroll 8
    for (int j = 0; j < 64; ++j) { float x = (float)As[tid][j]; s += x * x; }
    norms[tid] = 0.5f * s;
  }
  __syncthreads();

  size_t obase = ((size_t)bh * SEQ + row0) * N_FEAT;
  int n = lane & 15, half = lane >> 4;
#pragma unroll
  for (int it = 0; it < 8; ++it) {
    int tt = w + it * 8;             // 0..63
    int ti = tt >> 4, tj = tt & 15;  // ti: row-tile 0..3, tj: col-tile 0..15
    v8f acc = {};
#pragma unroll
    for (int ks = 0; ks < 2; ++ks) {
      v16bf a = frag_a_ld(&As[ti * 16][ks * 32], 64);
      v16bf bb = frag_b_ld(&Os[tj * 16][ks * 32], 64);
      acc = wmma_bf16(a, bb, acc);
    }
#pragma unroll
    for (int r = 0; r < 8; ++r) {
      int row = ti * 16 + half * 8 + r;
      float v = __expf(acc[r] - norms[row]) * 0.0625f;   // 1/sqrt(256)
      P[obase + (size_t)row * N_FEAT + tj * 16 + n] = (bf16)v;
    }
  }
}

// ---------------------------------------------------------------------------
// TDM: DMA one 64 x 256 bf16 tile (row stride 256 elems) into LDS.
// D# group0: count=1 | lds_addr | global_addr | type=2 ("image")
// D# group1: mask=0, data_size=1(2B), dims/strides/tile dims. 2D -> no g2/g3.
// Wave-scalar: caller must be inside a *scalar* (SGPR) branch; TDM ignores EXEC.
// ---------------------------------------------------------------------------
static __device__ __forceinline__ void tdm_load_tile_64x256(const bf16* g, void* l) {
  uint64_t ga = (uint64_t)(uintptr_t)g;
  v4u g0;
  g0[0] = 1u;                                      // count=1, user mode
  g0[1] = (uint32_t)(uintptr_t)l;                  // lds_addr
  g0[2] = (uint32_t)ga;                            // global_addr[31:0]
  g0[3] = (uint32_t)((ga >> 32) & 0x01FFFFFFu) | 0x80000000u;  // addr[56:32] | type=2
  v8u g1;
  g1[0] = 0x00010000u;                             // wg_mask=0, data_size=1 (2B)
  g1[1] = 256u << 16;                              // tensor_dim0 = 256
  g1[2] = 64u << 16;                               // tensor_dim0 hi=0, tensor_dim1 = 64
  g1[3] = 256u << 16;                              // tensor_dim1 hi=0, tile_dim0 = 256
  g1[4] = 64u;                                     // tile_dim1 = 64, tile_dim2 = 0
  g1[5] = 256u;                                    // tensor_dim0_stride = 256
  g1[6] = 0u;
  g1[7] = 0u;
  asm volatile("tensor_load_to_lds %0, %1" :: "s"(g0), "s"(g1) : "memory");
}

// ---------------------------------------------------------------------------
// Chunked causal linear attention. One block per (b,h); chunks of 64 steps.
// State kept transposed in LDS: St[d][m] (fp32), plus ksum[m].
// Qp/Kp chunk tiles are DMA'd into LDS by the Tensor Data Mover (wave 0)
// while the other waves stage V^T; TENSORcnt-synced before the barrier.
// ---------------------------------------------------------------------------
__global__ void __launch_bounds__(256)
k_causal(const bf16* __restrict__ Qp, const bf16* __restrict__ Kpg,
         const bf16* __restrict__ Vg, bf16* __restrict__ O) {
  __shared__ float St[64][256];    // 64 KB  state^T: [d][m]
  __shared__ float ksum[256];      // 1 KB
  __shared__ bf16 Qs[64][256];     // 32 KB
  __shared__ bf16 Ks[64][256];     // 32 KB
  __shared__ bf16 Vt[64][64];      // 8 KB   V^T: [d][c]
  __shared__ bf16 Am[64][64];      // 8 KB   masked intra scores
  __shared__ float Den[64];

  int tid = threadIdx.x, w = tid >> 5, lane = tid & 31;
  int n = lane & 15, half = lane >> 4;
  int bh = blockIdx.x;
  int b = bh >> 4, h = bh & 15;
  const bf16* qb = Qp + (size_t)bh * SEQ * N_FEAT;
  const bf16* kb = Kpg + (size_t)bh * SEQ * N_FEAT;

  int tt0 = w, tt1 = w + 8;                       // output tiles over [64 x 64]
  int ti0 = tt0 >> 2, tj0 = tt0 & 3, ti1 = tt1 >> 2, tj1 = tt1 & 3;

  for (int i = tid; i < 64 * 256; i += 256) ((float*)St)[i] = 0.f;
  ksum[tid] = 0.f;

  for (int ch = 0; ch < SEQ / 64; ++ch) {
    __syncthreads();
    // ---- load chunk: Qp/Kp via TDM (wave 0, scalar branch), V by all waves ----
    if (__builtin_amdgcn_readfirstlane((int)threadIdx.x) == 0) {
      tdm_load_tile_64x256(qb + (size_t)ch * 64 * N_FEAT, &Qs[0][0]);
      tdm_load_tile_64x256(kb + (size_t)ch * 64 * N_FEAT, &Ks[0][0]);
      __builtin_amdgcn_s_wait_tensorcnt(0);
    }
    for (int q = tid; q < 4096; q += 256) {       // V, transposed into Vt
      int c = q >> 6, d = q & 63;
      Vt[d][c] = Vg[((size_t)(b * SEQ + ch * 64 + c)) * D_MODEL + h * HEAD_DIM + d];
    }
    if (tid < 64) Den[tid] = EPSF;
    __syncthreads();

    // ---- phase A ----
    v8f accN0 = {}, accN1 = {};                   // inter+intra numerator tiles
    // GEMM1: Qp(chunk) @ S ; Bt = St ([d][m] fp32)
#pragma unroll
    for (int ks = 0; ks < 8; ++ks) {
      int k0 = ks * 32;
      v16bf a0 = frag_a_ld(&Qs[ti0 * 16][k0], 256);
      v16bf b0 = frag_b_ld_f32(&St[tj0 * 16][k0], 256);
      accN0 = wmma_bf16(a0, b0, accN0);
      v16bf a1 = frag_a_ld(&Qs[ti1 * 16][k0], 256);
      v16bf b1 = frag_b_ld_f32(&St[tj1 * 16][k0], 256);
      accN1 = wmma_bf16(a1, b1, accN1);
    }
    // den_inter = Qp @ ksum
    if (tid < 64) {
      float s = 0.f;
      for (int m = 0; m < 256; ++m) s += (float)Qs[tid][m] * ksum[m];
      atomicAdd(&Den[tid], s);
    }
    // GEMM2: Scores = Qp @ Kp^T ; Bt = Ks row-major -> mask -> Am
    {
      v8f s0 = {}, s1 = {};
#pragma unroll
      for (int ks = 0; ks < 8; ++ks) {
        int k0 = ks * 32;
        v16bf a0 = frag_a_ld(&Qs[ti0 * 16][k0], 256);
        v16bf b0 = frag_b_ld(&Ks[tj0 * 16][k0], 256);
        s0 = wmma_bf16(a0, b0, s0);
        v16bf a1 = frag_a_ld(&Qs[ti1 * 16][k0], 256);
        v16bf b1 = frag_b_ld(&Ks[tj1 * 16][k0], 256);
        s1 = wmma_bf16(a1, b1, s1);
      }
#pragma unroll
      for (int t = 0; t < 2; ++t) {
        int ti = t ? ti1 : ti0, tj = t ? tj1 : tj0;
        v8f s = t ? s1 : s0;
#pragma unroll
        for (int r = 0; r < 8; ++r) {
          int row = ti * 16 + half * 8 + r;
          int col = tj * 16 + n;
          float v = (col <= row) ? s[r] : 0.f;    // causal incl. diagonal
          Am[row][col] = (bf16)v;
        }
      }
    }
    __syncthreads();

    // ---- phase B ----
    // GEMM3: N += Am @ V ; Bt = Vt
#pragma unroll
    for (int ks = 0; ks < 2; ++ks) {
      int k0 = ks * 32;
      v16bf a0 = frag_a_ld(&Am[ti0 * 16][k0], 64);
      v16bf b0 = frag_b_ld(&Vt[tj0 * 16][k0], 64);
      accN0 = wmma_bf16(a0, b0, accN0);
      v16bf a1 = frag_a_ld(&Am[ti1 * 16][k0], 64);
      v16bf b1 = frag_b_ld(&Vt[tj1 * 16][k0], 64);
      accN1 = wmma_bf16(a1, b1, accN1);
    }
    // GEMM4: St += V^T @ Kp  (out [d][m]: 4 x 16 tiles, 8 per wave)
#pragma unroll
    for (int it = 0; it < 8; ++it) {
      int tt = w + it * 8;
      int ti = tt >> 4, tj = tt & 15;
      v8f c;
#pragma unroll
      for (int r = 0; r < 8; ++r) c[r] = St[ti * 16 + half * 8 + r][tj * 16 + n];
#pragma unroll
      for (int ks = 0; ks < 2; ++ks) {
        int k0 = ks * 32;
        v16bf a = frag_a_ld(&Vt[ti * 16][k0], 64);
        v16bf bb;                                  // Bt[n][k] = Ks[k][m] (strided)
        int mcol = tj * 16 + n;
#pragma unroll
        for (int i = 0; i < 16; ++i) bb[i] = Ks[k0 + half * 16 + i][mcol];
        c = wmma_bf16(a, bb, c);
      }
#pragma unroll
      for (int r = 0; r < 8; ++r) St[ti * 16 + half * 8 + r][tj * 16 + n] = c[r];
    }
    // ksum += colsum(Kp)
    {
      float s = 0.f;
#pragma unroll 8
      for (int c = 0; c < 64; ++c) s += (float)Ks[c][tid];
      ksum[tid] += s;
    }
    // den += rowsum(Am) (already masked)
    if (tid < 64) {
      float s = 0.f;
#pragma unroll 8
      for (int j = 0; j < 64; ++j) s += (float)Am[tid][j];
      Den[tid] += s;
    }
    __syncthreads();

    // ---- epilogue: O = N / den ----
    size_t ob = ((size_t)(b * SEQ + ch * 64)) * D_MODEL + h * HEAD_DIM;
#pragma unroll
    for (int t = 0; t < 2; ++t) {
      int ti = t ? ti1 : ti0, tj = t ? tj1 : tj0;
      v8f acc = t ? accN1 : accN0;
#pragma unroll
      for (int r = 0; r < 8; ++r) {
        int row = ti * 16 + half * 8 + r;
        int col = tj * 16 + n;
        O[ob + (size_t)row * D_MODEL + col] = (bf16)(acc[r] / Den[row]);
      }
    }
  }
}

// ---------------------------------------------------------------------------
extern "C" void kernel_launch(void* const* d_in, const int* in_sizes, int n_in,
                              void* d_out, int out_size, void* d_ws, size_t ws_size,
                              hipStream_t stream) {
  (void)in_sizes; (void)n_in; (void)out_size; (void)ws_size;
  const float* x  = (const float*)d_in[0];
  const float* Wq = (const float*)d_in[1];
  const float* bq = (const float*)d_in[2];
  const float* Wk = (const float*)d_in[3];
  const float* bk = (const float*)d_in[4];
  const float* Wv = (const float*)d_in[5];
  const float* bv = (const float*)d_in[6];
  const float* Wo = (const float*)d_in[7];
  const float* bo = (const float*)d_in[8];
  const float* om = (const float*)d_in[9];
  float* out = (float*)d_out;

  char* ws = (char*)d_ws;
  size_t off = 0;
  auto take = [&](size_t bytes) -> char* {
    char* p = ws + off;
    off += (bytes + 255) & ~(size_t)255;
    return p;
  };
  bf16* xb  = (bf16*)take((size_t)ROWS * D_MODEL * 2);          // 32 MB
  bf16* WqT = (bf16*)take((size_t)D_MODEL * D_MODEL * 2);       // 2 MB
  bf16* WkT = (bf16*)take((size_t)D_MODEL * D_MODEL * 2);
  bf16* WvT = (bf16*)take((size_t)D_MODEL * D_MODEL * 2);
  bf16* WoT = (bf16*)take((size_t)D_MODEL * D_MODEL * 2);
  bf16* omb = (bf16*)take((size_t)N_FEAT * HEAD_DIM * 2);
  bf16* Qsc = (bf16*)take((size_t)ROWS * D_MODEL * 2);          // scaled q
  bf16* Ksc = (bf16*)take((size_t)ROWS * D_MODEL * 2);          // scaled k
  bf16* Vb  = (bf16*)take((size_t)ROWS * D_MODEL * 2);
  bf16* qp  = (bf16*)take((size_t)BATCH * N_HEADS * SEQ * N_FEAT * 2);  // 128 MB
  bf16* kp  = (bf16*)take((size_t)BATCH * N_HEADS * SEQ * N_FEAT * 2);  // 128 MB
  bf16* Ob  = (bf16*)take((size_t)ROWS * D_MODEL * 2);

  // conversions
  {
    size_t nx = (size_t)ROWS * D_MODEL;
    k_cvt<<<(unsigned)((nx + 255) / 256), 256, 0, stream>>>(x, xb, nx);
  }
  {
    unsigned g = (D_MODEL * D_MODEL + 255) / 256;
    k_transpose_cvt<<<g, 256, 0, stream>>>(Wq, WqT, D_MODEL, D_MODEL);
    k_transpose_cvt<<<g, 256, 0, stream>>>(Wk, WkT, D_MODEL, D_MODEL);
    k_transpose_cvt<<<g, 256, 0, stream>>>(Wv, WvT, D_MODEL, D_MODEL);
    k_transpose_cvt<<<g, 256, 0, stream>>>(Wo, WoT, D_MODEL, D_MODEL);
  }
  k_cvt<<<(N_FEAT * HEAD_DIM + 255) / 256, 256, 0, stream>>>(
      om, omb, (size_t)N_FEAT * HEAD_DIM);

  // projections (q,k pre-scaled by HEAD_DIM^-0.25 = 2^-1.5)
  dim3 gg(D_MODEL / 64, ROWS / 64);
  const float s = 0.35355339059327373f;
  k_gemm<<<gg, 256, 0, stream>>>(xb, WqT, bq, nullptr, Qsc, ROWS, D_MODEL, D_MODEL, s);
  k_gemm<<<gg, 256, 0, stream>>>(xb, WkT, bk, nullptr, Ksc, ROWS, D_MODEL, D_MODEL, s);
  k_gemm<<<gg, 256, 0, stream>>>(xb, WvT, bv, nullptr, Vb, ROWS, D_MODEL, D_MODEL, 1.f);

  // feature maps
  dim3 gp(SEQ / 64, BATCH * N_HEADS);
  k_phi<<<gp, 256, 0, stream>>>(Qsc, omb, qp);
  k_phi<<<gp, 256, 0, stream>>>(Ksc, omb, kp);

  // chunked causal linear attention (TDM-fed)
  k_causal<<<BATCH * N_HEADS, 256, 0, stream>>>(qp, kp, Vb, Ob);

  // output projection (fp32 out)
  k_gemm<<<gg, 256, 0, stream>>>(Ob, WoT, bo, out, nullptr, ROWS, D_MODEL, D_MODEL, 1.f);
}
